// ProtoNet_41867341201597
// MI455X (gfx1250) — compile-verified, matrix operand order checked by reading, not run on hardware
//
#include <hip/hip_runtime.h>
#include <hip/hip_bf16.h>
#include <stdint.h>

typedef __attribute__((ext_vector_type(16))) __bf16 v16bf;
typedef __attribute__((ext_vector_type(8)))  float  v8f;

union Frag {
    v16bf v;
    uint4 q[2];
    unsigned short s[16];
};

__device__ __forceinline__ unsigned short f2bf(float f) {
    unsigned int u = __float_as_uint(f);
    u += 0x7FFFu + ((u >> 16) & 1u);      // round-to-nearest-even
    return (unsigned short)(u >> 16);
}

// ---------------------------------------------------------------------------
// Kernel 1: W [2048,512] f32 row-major -> Wt [512][2048] bf16 (transposed)
// thread t: n = t%512, kc = t/512 handles 8 consecutive k -> 16B packed store
// ---------------------------------------------------------------------------
__global__ __launch_bounds__(256) void wt_kernel(const float* __restrict__ W,
                                                 unsigned short* __restrict__ Wt) {
    int id = blockIdx.x * blockDim.x + threadIdx.x;   // 512*256 = 131072 threads
    int kc = id >> 9;            // 0..255
    int n  = id & 511;
    union { unsigned short s[8]; uint4 q; } pk;
#pragma unroll
    for (int j = 0; j < 8; ++j)
        pk.s[j] = f2bf(W[(size_t)(kc * 8 + j) * 512 + n]);
    *(uint4*)(Wt + (size_t)n * 2048 + kc * 8) = pk.q;
}

// ---------------------------------------------------------------------------
// Kernel 2: Z = X @ W + b via v_wmma_f32_16x16x32_bf16
// block = 256 thr (8 waves). Block owns 32 rows; A tile staged bf16 in LDS
// (32 x 2048, ~128.5 KB with padding — fits gfx1250's 320 KB LDS).
// Wave w owns cols [w*64, w*64+64): 8 accumulators (2 M-tiles x 4 N-tiles),
// each B fragment loaded once and used by 2 WMMAs.
// ---------------------------------------------------------------------------
#define LDS_STRIDE 2056   // 2048 + 8 ushort pad: rows land in distinct banks
#define MTILE 32

__global__ __launch_bounds__(256) void gemm_bias_kernel(
        const float* __restrict__ X, const unsigned short* __restrict__ Wt,
        const float* __restrict__ bias, float* __restrict__ Z) {
    __shared__ unsigned short As[MTILE * LDS_STRIDE];

    const int tid = threadIdx.x;
    const int m0  = blockIdx.x * MTILE;

    // cooperative load: 32x2048 f32 -> bf16 LDS (64 float4 per thread)
#pragma unroll
    for (int j = 0; j < 64; ++j) {
        int e4   = tid + 256 * j;        // float4 index 0..16383
        int elem = e4 * 4;
        int row  = elem >> 11;
        int col  = elem & 2047;
        float4 f = *(const float4*)(X + (size_t)(m0 + row) * 2048 + col);
        union { unsigned short s[4]; uint2 u; } pk;
        pk.s[0] = f2bf(f.x); pk.s[1] = f2bf(f.y);
        pk.s[2] = f2bf(f.z); pk.s[3] = f2bf(f.w);
        *(uint2*)(As + row * LDS_STRIDE + col) = pk.u;
    }
    __syncthreads();

    const int wave  = tid >> 5;
    const int lane  = tid & 31;
    const int row   = lane & 15;
    const int khalf = (lane >> 4) << 3;      // 0 or 8
    const int nbase = wave * 64;

    v8f acc[8] = {};
    for (int k0 = 0; k0 < 2048; k0 += 32) {
        Frag a0, a1;
        const unsigned short* ap0 = As + row * LDS_STRIDE + k0 + khalf;
        const unsigned short* ap1 = ap0 + 16 * LDS_STRIDE;
        a0.q[0] = *(const uint4*)ap0;          // rows 0..15,  K = khalf..+7
        a0.q[1] = *(const uint4*)(ap0 + 16);   //              K = khalf+16..+23
        a1.q[0] = *(const uint4*)ap1;          // rows 16..31
        a1.q[1] = *(const uint4*)(ap1 + 16);
#pragma unroll
        for (int nt = 0; nt < 4; ++nt) {
            Frag bfr;
            const unsigned short* bp =
                Wt + (size_t)(nbase + nt * 16 + row) * 2048 + k0 + khalf;
            bfr.q[0] = *(const uint4*)bp;
            bfr.q[1] = *(const uint4*)(bp + 16);
            acc[nt] = __builtin_amdgcn_wmma_f32_16x16x32_bf16(
                false, a0.v, false, bfr.v, (short)0, acc[nt], false, false);
            acc[4 + nt] = __builtin_amdgcn_wmma_f32_16x16x32_bf16(
                false, a1.v, false, bfr.v, (short)0, acc[4 + nt], false, false);
        }
    }

    const int mofs = (lane >> 4) * 8;   // lanes 16..31 hold rows M+8
#pragma unroll
    for (int mt = 0; mt < 2; ++mt) {
#pragma unroll
        for (int nt = 0; nt < 4; ++nt) {
            int n = nbase + nt * 16 + (lane & 15);
            float bv = bias[n];
#pragma unroll
            for (int r = 0; r < 8; ++r) {
                int m = m0 + mt * 16 + r + mofs;
                Z[(size_t)m * 512 + n] = acc[mt * 4 + nt][r] + bv;
            }
        }
    }
}

// ---------------------------------------------------------------------------
// Kernel 3: deterministic prototypes. Block n scans all S=1024 support rows.
// Emits bf16 prototype row + squared norm.
// ---------------------------------------------------------------------------
__global__ __launch_bounds__(256) void proto_kernel(
        const float* __restrict__ Zs, const int* __restrict__ ys,
        unsigned short* __restrict__ protoB, float* __restrict__ pnorm) {
    const int n   = blockIdx.x;        // 0..63
    const int tid = threadIdx.x;       // 2 columns per thread
    const int d0  = tid * 2;
    float acc0 = 0.f, acc1 = 0.f, cnt = 0.f;
    for (int s = 0; s < 1024; ++s) {
        if (ys[s] == n) {
            cnt  += 1.f;
            acc0 += Zs[(size_t)s * 512 + d0];
            acc1 += Zs[(size_t)s * 512 + d0 + 1];
        }
    }
    float inv = 1.f / fmaxf(cnt, 1.f);
    float p0 = acc0 * inv, p1 = acc1 * inv;
    union { unsigned short s[2]; unsigned int u; } pk;
    pk.s[0] = f2bf(p0); pk.s[1] = f2bf(p1);
    *(unsigned int*)(protoB + (size_t)n * 512 + d0) = pk.u;

    __shared__ float red[256];
    red[tid] = p0 * p0 + p1 * p1;
    __syncthreads();
    for (int off = 128; off > 0; off >>= 1) {
        if (tid < off) red[tid] += red[tid + off];
        __syncthreads();
    }
    if (tid == 0) pnorm[n] = red[0];
}

// ---------------------------------------------------------------------------
// Kernel 4: per-query squared norms, one wave32 per row, shuffle reduction
// ---------------------------------------------------------------------------
__global__ __launch_bounds__(256) void qnorm_kernel(const float* __restrict__ Zq,
                                                    float* __restrict__ qn) {
    int wid  = (blockIdx.x * blockDim.x + threadIdx.x) >> 5;   // row id
    int lane = threadIdx.x & 31;
    const float* rowp = Zq + (size_t)wid * 512;
    float acc = 0.f;
#pragma unroll
    for (int j = 0; j < 16; ++j) {
        float v = rowp[lane + 32 * j];
        acc += v * v;
    }
#pragma unroll
    for (int m = 16; m >= 1; m >>= 1) acc += __shfl_xor(acc, m, 32);
    if (lane == 0) qn[wid] = acc;
}

// ---------------------------------------------------------------------------
// Kernel 5: out[q,n] = -(||q||^2 + ||p||^2 - 2 q.p), q.p via WMMA bf16.
// Wave owns 16 query rows x all 64 classes (4 accumulators). K = 512.
// protoB (64 KB) stays L1/L2 resident; dominant traffic is the Zq stream.
// ---------------------------------------------------------------------------
__global__ __launch_bounds__(256) void dist_kernel(
        const float* __restrict__ Zq, const unsigned short* __restrict__ protoB,
        const float* __restrict__ qn, const float* __restrict__ pnorm,
        float* __restrict__ out) {
    const int tid   = threadIdx.x;
    const int wave  = tid >> 5;
    const int lane  = tid & 31;
    const int q0    = (blockIdx.x * 8 + wave) * 16;
    const int row   = lane & 15;
    const int khalf = (lane >> 4) << 3;

    v8f acc[4] = {};
    for (int k0 = 0; k0 < 512; k0 += 32) {
        Frag a;
        const float* ap = Zq + (size_t)(q0 + row) * 512 + k0 + khalf;
        float tf[16];
        *(float4*)(tf + 0)  = *(const float4*)(ap);
        *(float4*)(tf + 4)  = *(const float4*)(ap + 4);
        *(float4*)(tf + 8)  = *(const float4*)(ap + 16);
        *(float4*)(tf + 12) = *(const float4*)(ap + 20);
#pragma unroll
        for (int j = 0; j < 16; ++j) a.s[j] = f2bf(tf[j]);
#pragma unroll
        for (int nt = 0; nt < 4; ++nt) {
            Frag bfr;
            const unsigned short* bp =
                protoB + (size_t)(nt * 16 + row) * 512 + k0 + khalf;
            bfr.q[0] = *(const uint4*)bp;
            bfr.q[1] = *(const uint4*)(bp + 16);
            acc[nt] = __builtin_amdgcn_wmma_f32_16x16x32_bf16(
                false, a.v, false, bfr.v, (short)0, acc[nt], false, false);
        }
    }

    const int mofs = (lane >> 4) * 8;
#pragma unroll
    for (int nt = 0; nt < 4; ++nt) {
        int n = nt * 16 + (lane & 15);
        float pn = pnorm[n];
#pragma unroll
        for (int r = 0; r < 8; ++r) {
            int q = q0 + r + mofs;
            out[(size_t)q * 64 + n] = 2.f * acc[nt][r] - qn[q] - pn;
        }
    }
}

// ---------------------------------------------------------------------------
extern "C" void kernel_launch(void* const* d_in, const int* in_sizes, int n_in,
                              void* d_out, int out_size, void* d_ws, size_t ws_size,
                              hipStream_t stream) {
    const float* xs = (const float*)d_in[0];   // [1024, 2048]
    const int*   ys = (const int*)d_in[1];     // [1024]
    const float* xq = (const float*)d_in[2];   // [16384, 2048]
    const float* W  = (const float*)d_in[3];   // [2048, 512]
    const float* b  = (const float*)d_in[4];   // [512]
    float* out = (float*)d_out;                // [16384, 64]

    char* ws = (char*)d_ws;
    unsigned short* Wt     = (unsigned short*)(ws);                          // 2 MB
    float*          Zs     = (float*)(ws + (size_t)(2  << 20));              // 2 MB
    float*          Zq     = (float*)(ws + (size_t)(4  << 20));              // 32 MB
    unsigned short* protoB = (unsigned short*)(ws + (size_t)(36 << 20));     // 64 KB
    float*          pnorm  = (float*)(ws + (size_t)(36 << 20) + (64 << 10)); // 256 B
    float*          qn     = (float*)(ws + (size_t)(36 << 20) + (128 << 10));// 64 KB

    wt_kernel       <<<512,  256, 0, stream>>>(W, Wt);
    gemm_bias_kernel<<<32,   256, 0, stream>>>(xs, Wt, b, Zs);   // M = 1024
    gemm_bias_kernel<<<512,  256, 0, stream>>>(xq, Wt, b, Zq);   // M = 16384
    proto_kernel    <<<64,   256, 0, stream>>>(Zs, ys, protoB, pnorm);
    qnorm_kernel    <<<2048, 256, 0, stream>>>(Zq, qn);
    dist_kernel     <<<128,  256, 0, stream>>>(Zq, protoB, qn, pnorm, out);
}
